// MoChA_34437047779602
// MI455X (gfx1250) — matrix operand order repaired and checked by reading, not compiled, not run on hardware
//
#include <hip/hip_runtime.h>

// ---------------------------------------------------------------------------
// MoChA beta on gfx1250: bf16 WMMA for all GEMMs, fp32 accumulate/elementwise.
// Shapes: B=8, qlen=256, klen=2048, adim=512, H=4, dk=128, CHUNK=4.
// Round 2: all fp32->bf16 conversion hoisted out of GEMM inner loops; weight
// matrices pre-transposed to bf16 so every WMMA fragment is a contiguous
// 16/32-byte global load.
// ---------------------------------------------------------------------------

namespace {
constexpr int   kB  = 8;
constexpr int   kQ  = 256;
constexpr int   kK  = 2048;
constexpr int   kA  = 512;
constexpr int   kH  = 4;
constexpr int   kDK = 128;
constexpr float kInvScale = 0.044194173824159216f;  // 1/sqrt(512)
constexpr float kEps      = 1e-6f;
}

typedef __attribute__((ext_vector_type(16))) __bf16 v16bf;
typedef __attribute__((ext_vector_type(8)))  float  v8f;

union V16 {
    uint4          q[2];
    unsigned short s[16];
    v16bf          v;
};

// float -> bf16, round-to-nearest-even
__device__ __forceinline__ unsigned short f2bf(float f) {
    unsigned int u = __float_as_uint(f);
    u += 0x7FFFu + ((u >> 16) & 1u);
    return (unsigned short)(u >> 16);
}

// A-matrix fragment (16x32 bf16): lane holds row M=lane&15; per ISA layout the
// 16 elements are two contiguous runs of 8: K in [kc+hf*8, +8) and [kc+16+hf*8, +8).
__device__ __forceinline__ v16bf frag_a(const unsigned short* __restrict__ row,
                                        int kc, int hf) {
    V16 f;
    f.q[0] = *(const uint4*)(row + kc + hf * 8);
    f.q[1] = *(const uint4*)(row + kc + 16 + hf * 8);
    return f.v;
}

// B-matrix fragment (32x16 bf16): lane holds column N=lane&15; same K order as A:
// two contiguous runs of 8: K in [kc+hf*8, +8) and [kc+16+hf*8, +8).
__device__ __forceinline__ v16bf frag_b(const unsigned short* __restrict__ row,
                                        int kc, int hf) {
    V16 f;
    f.q[0] = *(const uint4*)(row + kc + hf * 8);
    f.q[1] = *(const uint4*)(row + kc + 16 + hf * 8);
    return f.v;
}

// ---------------------------------------------------------------------------
// Kernel 0a: streaming fp32 -> bf16 conversion (float4 in, ushort4 out).
// ---------------------------------------------------------------------------
__global__ __launch_bounds__(256) void cvt_bf16(
        const float* __restrict__ src, unsigned short* __restrict__ dst, int n4) {
    const int i = blockIdx.x * blockDim.x + threadIdx.x;
    if (i >= n4) return;
    const float4 v = ((const float4*)src)[i];
    ushort4 o;
    o.x = f2bf(v.x); o.y = f2bf(v.y); o.z = f2bf(v.z); o.w = f2bf(v.w);
    ((ushort4*)dst)[i] = o;
}

// ---------------------------------------------------------------------------
// Kernel 0b: W[512,512] fp32 -> Wt[512,512] bf16 transposed (Wt[n][k]=W[k][n]).
// 32x32 LDS tiles, 256 blocks of 256 threads.
// ---------------------------------------------------------------------------
__global__ __launch_bounds__(256) void wtr_bf16(
        const float* __restrict__ W, unsigned short* __restrict__ Wt) {
    __shared__ float tile[32][33];
    const int c0 = (blockIdx.x & 15) * 32;
    const int r0 = (blockIdx.x >> 4) * 32;
    const int tc = threadIdx.x & 31;
    const int tr = threadIdx.x >> 5;  // 0..7
    for (int rr = tr; rr < 32; rr += 8)
        tile[rr][tc] = W[(size_t)(r0 + rr) * kA + c0 + tc];
    __syncthreads();
    for (int rr = tr; rr < 32; rr += 8)
        Wt[(size_t)(c0 + rr) * kA + r0 + tc] = f2bf(tile[tc][rr]);
}

// ---------------------------------------------------------------------------
// Kernel 1: C[M,512] = A[M,512] @ W[512,512] + bias, all-bf16 operands
// (A pre-converted, W pre-transposed), bf16 output. One 16x16 C-tile per wave.
// ---------------------------------------------------------------------------
__global__ __launch_bounds__(128) void proj_bf16_wmma(
        const unsigned short* __restrict__ A, const unsigned short* __restrict__ Wt,
        const float* __restrict__ bias, unsigned short* __restrict__ C, int M) {
    const int N = kA;
    const int tiles = (M / 16) * (N / 16);
    const int w    = threadIdx.x >> 5;
    const int lane = threadIdx.x & 31;
    const int t    = blockIdx.x * 4 + w;
    if (t >= tiles) return;                       // wave-uniform
    const int mt  = t / (N / 16);
    const int nt  = t % (N / 16);
    const int hf  = lane >> 4;
    const int l15 = lane & 15;
    const int m_row = mt * 16 + l15;
    const int n     = nt * 16 + l15;

    const unsigned short* arow = A + (size_t)m_row * kA;
    const unsigned short* brow = Wt + (size_t)n * kA;   // column n of W, contiguous
    v8f acc = {};
    for (int kc = 0; kc < kA; kc += 32) {
        v16bf af = frag_a(arow, kc, hf);
        v16bf bf = frag_b(brow, kc, hf);
        acc = __builtin_amdgcn_wmma_f32_16x16x32_bf16(
                false, af, false, bf, (short)0, acc, false, false);
    }
    const float bv = bias[n];
#pragma unroll
    for (int rr = 0; rr < 8; ++rr) {
        const int m = mt * 16 + rr + hf * 8;      // D: VGPR rr -> M = rr + hf*8
        C[(size_t)m * kA + n] = f2bf(acc[rr] + bv);
    }
}

// ---------------------------------------------------------------------------
// Kernel 2: monotonic energy tile -> p_choose, exclusive cumprod(1-p).
// One block per (b, 16-query tile). 256 threads = 8 waves.
// Dynamic LDS: 16*2048 floats (128 KB energy tile, reused for log1mp).
// ---------------------------------------------------------------------------
__global__ __launch_bounds__(256) void mono_pchoose_wmma(
        const unsigned short* __restrict__ qm, const unsigned short* __restrict__ km,
        const float* __restrict__ noise, const float* __restrict__ rptr,
        float* __restrict__ p_buf, float* __restrict__ cp_buf) {
    extern __shared__ float etile[];              // [16][2048]
    const int b   = blockIdx.x >> 4;
    const int q0  = (blockIdx.x & 15) * 16;
    const int w   = threadIdx.x >> 5;
    const int lane = threadIdx.x & 31;
    const int hf  = lane >> 4;
    const int l15 = lane & 15;
    const float rv = rptr[0];

    const unsigned short* arow = qm + (size_t)(b * kQ + q0 + l15) * kA;

    for (int nt = w; nt < kK / 16; nt += 8) {     // wave-uniform -> EXEC all 1s
        const unsigned short* brow = km + (size_t)(b * kK + nt * 16 + l15) * kA;
        v8f acc = {};
        for (int kc = 0; kc < kA; kc += 32) {
            v16bf af = frag_a(arow, kc, hf);
            v16bf bf = frag_b(brow, kc, hf);
            acc = __builtin_amdgcn_wmma_f32_16x16x32_bf16(
                    false, af, false, bf, (short)0, acc, false, false);
        }
#pragma unroll
        for (int rr = 0; rr < 8; ++rr)
            etile[(rr + hf * 8) * kK + nt * 16 + l15] = acc[rr] * kInvScale + rv;
    }
    __syncthreads();

    // p = sigmoid(e + noise); tile <- log(clamp(1-p, eps, 1))
    const size_t gbase = (size_t)(b * kQ + q0) * kK;
    for (int idx = threadIdx.x; idx < 16 * kK; idx += blockDim.x) {
        const int m = idx >> 11;                  // kK == 2048
        const int k = idx & (kK - 1);
        const float e = etile[idx] + noise[gbase + (size_t)m * kK + k];
        const float p = 1.0f / (1.0f + expf(-e));
        p_buf[gbase + (size_t)m * kK + k] = p;
        etile[idx] = logf(fmaxf(1.0f - p, kEps));
    }
    __syncthreads();

    // per-row exclusive cumsum of log1mp -> cp = exp(prefix). One wave per row.
    for (int row = w; row < 16; row += 8) {
        const float* src = etile + row * kK;
        const int base = lane * 64;               // 2048 / 32 lanes
        float partial = 0.f;
        for (int j = 0; j < 64; ++j) partial += src[base + j];
        float x = partial;
        for (int d = 1; d < 32; d <<= 1) {
            const float y = __shfl_up(x, d);
            if (lane >= d) x += y;
        }
        float run = x - partial;                  // exclusive prefix
        float* dst = cp_buf + gbase + (size_t)row * kK;
        for (int j = 0; j < 64; ++j) {
            dst[base + j] = expf(run);
            run += src[base + j];
        }
    }
}

// ---------------------------------------------------------------------------
// Kernel 3: MoChA alpha recurrence, sequential over 256 query steps.
// One block per batch; 512 threads each own 4 contiguous k-elements.
// ---------------------------------------------------------------------------
__global__ __launch_bounds__(512) void alpha_scan(
        const float* __restrict__ p_buf, const float* __restrict__ cp_buf,
        float* __restrict__ alpha_buf) {
    __shared__ float wsum[16];
    __shared__ float wexc[16];
    const int b    = blockIdx.x;
    const int tid  = threadIdx.x;
    const int w    = tid >> 5;
    const int lane = tid & 31;
    const int base = tid * 4;

    float aw[4];
#pragma unroll
    for (int j = 0; j < 4; ++j) aw[j] = (base + j == 0) ? 1.0f : 0.0f;

    for (int q = 0; q < kQ; ++q) {
        const size_t roff = (size_t)(b * kQ + q) * kK + base;
        if (q + 1 < kQ) {                         // global_prefetch_b8 next row
            __builtin_prefetch(p_buf + roff + kK, 0, 1);
            __builtin_prefetch(cp_buf + roff + kK, 0, 1);
        }
        const float4 p4 = *(const float4*)(p_buf + roff);
        const float4 c4 = *(const float4*)(cp_buf + roff);
        const float p[4]  = {p4.x, p4.y, p4.z, p4.w};
        const float cp[4] = {c4.x, c4.y, c4.z, c4.w};
        float s[4], tot = 0.f;
#pragma unroll
        for (int j = 0; j < 4; ++j) {
            s[j] = aw[j] / fmaxf(cp[j], kEps);
            tot += s[j];
        }
        float x = tot;                            // wave inclusive scan
        for (int d = 1; d < 32; d <<= 1) {
            const float y = __shfl_up(x, d);
            if (lane >= d) x += y;
        }
        if (lane == 31) wsum[w] = x;
        const float texc = x - tot;
        __syncthreads();
        if (tid < 16) {                           // scan the 16 wave totals
            const float v = wsum[tid];
            float xx = v;
            for (int d = 1; d < 16; d <<= 1) {
                const float y = __shfl_up(xx, d);
                if (tid >= d) xx += y;
            }
            wexc[tid] = xx - v;
        }
        __syncthreads();
        float run = wexc[w] + texc;
        float4 o;
        float ov[4];
#pragma unroll
        for (int j = 0; j < 4; ++j) {
            run += s[j];
            ov[j] = p[j] * cp[j] * run;
            aw[j] = ov[j];
        }
        o.x = ov[0]; o.y = ov[1]; o.z = ov[2]; o.w = ov[3];
        *(float4*)(alpha_buf + roff) = o;
        __syncthreads();                          // protect wsum for next step
    }
}

// ---------------------------------------------------------------------------
// Kernel 4: chunkwise energies (WMMA, dk=128) + fused moving-sum beta.
// One block per (b, h, 16-query tile). Dynamic LDS: 16*2048 f32 + 16 f32.
// ---------------------------------------------------------------------------
__global__ __launch_bounds__(256) void chunk_beta_wmma(
        const unsigned short* __restrict__ qc, const unsigned short* __restrict__ kcb,
        const float* __restrict__ alpha_buf, float* __restrict__ out) {
    extern __shared__ float smem[];
    float* etile  = smem;                         // [16][2048]
    float* rowmax = smem + 16 * kK;               // [16]
    const int bh  = blockIdx.x >> 4;
    const int q0  = (blockIdx.x & 15) * 16;
    const int b   = bh >> 2;
    const int h   = bh & 3;
    const int w   = threadIdx.x >> 5;
    const int lane = threadIdx.x & 31;
    const int hf  = lane >> 4;
    const int l15 = lane & 15;

    const unsigned short* arow = qc + (size_t)(b * kQ + q0 + l15) * kA + h * kDK;

    for (int nt = w; nt < kK / 16; nt += 8) {     // wave-uniform
        const unsigned short* brow =
                kcb + (size_t)(b * kK + nt * 16 + l15) * kA + h * kDK;
        v8f acc = {};
#pragma unroll
        for (int kc = 0; kc < kDK; kc += 32) {
            v16bf af = frag_a(arow, kc, hf);
            v16bf bf = frag_b(brow, kc, hf);
            acc = __builtin_amdgcn_wmma_f32_16x16x32_bf16(
                    false, af, false, bf, (short)0, acc, false, false);
        }
#pragma unroll
        for (int rr = 0; rr < 8; ++rr)
            etile[(rr + hf * 8) * kK + nt * 16 + l15] = acc[rr] * kInvScale;
    }
    __syncthreads();

    for (int row = w; row < 16; row += 8) {       // row max
        float m = -3.4e38f;
        for (int k = lane; k < kK; k += 32) m = fmaxf(m, etile[row * kK + k]);
#pragma unroll
        for (int d = 16; d >= 1; d >>= 1) m = fmaxf(m, __shfl_xor(m, d));
        if (lane == 0) rowmax[row] = m;
    }
    __syncthreads();

    for (int idx = threadIdx.x; idx < 16 * kK; idx += blockDim.x) {
        const int row = idx >> 11;
        etile[idx] = fmaxf(expf(etile[idx] - rowmax[row]), 1e-5f);
    }
    __syncthreads();

    // beta[k] = sexp[k] * sum_{j=k..k+3} alpha[j] / (sexp[j-3]+...+sexp[j])
    for (int row = w; row < 16; row += 8) {
        const int q = q0 + row;
        const float* sx   = etile + row * kK;
        const float* al   = alpha_buf + (size_t)(b * kQ + q) * kK;
        float*       orow = out + ((size_t)(b * kH + h) * kQ + q) * kK;
        for (int k = lane; k < kK; k += 32) {
            float sum = 0.f;
            const int jmax = (k + 3 < kK - 1) ? k + 3 : kK - 1;
            for (int j = k; j <= jmax; ++j) {
                float den = 0.f;
                const int i0 = (j - 3 > 0) ? j - 3 : 0;
                for (int i = i0; i <= j; ++i) den += sx[i];
                sum += al[j] / den;
            }
            orow[k] = sx[k] * sum;
        }
    }
}

// ---------------------------------------------------------------------------
extern "C" void kernel_launch(void* const* d_in, const int* in_sizes, int n_in,
                              void* d_out, int out_size, void* d_ws, size_t ws_size,
                              hipStream_t stream) {
    (void)in_sizes; (void)n_in; (void)out_size; (void)ws_size;
    const float* key_enc = (const float*)d_in[0];
    const float* query   = (const float*)d_in[1];
    const float* noise   = (const float*)d_in[2];
    const float* Wk_m    = (const float*)d_in[3];
    const float* bk_m    = (const float*)d_in[4];
    const float* Wq_m    = (const float*)d_in[5];
    const float* bq_m    = (const float*)d_in[6];
    const float* rscal   = (const float*)d_in[7];
    const float* Wk_c    = (const float*)d_in[8];
    const float* bk_c    = (const float*)d_in[9];
    const float* Wq_c    = (const float*)d_in[10];
    const float* bq_c    = (const float*)d_in[11];
    float* out = (float*)d_out;

    char* ws = (char*)d_ws;
    size_t off = 0;
    auto take = [&](size_t bytes) -> void* {
        void* p = ws + off;
        off += (bytes + 255) & ~(size_t)255;
        return p;
    };
    unsigned short* km    = (unsigned short*)take((size_t)kB * kK * kA * 2);
    unsigned short* kc    = (unsigned short*)take((size_t)kB * kK * kA * 2);
    unsigned short* qm    = (unsigned short*)take((size_t)kB * kQ * kA * 2);
    unsigned short* qc    = (unsigned short*)take((size_t)kB * kQ * kA * 2);
    unsigned short* keyb  = (unsigned short*)take((size_t)kB * kK * kA * 2);
    unsigned short* qryb  = (unsigned short*)take((size_t)kB * kQ * kA * 2);
    unsigned short* wkm_t = (unsigned short*)take((size_t)kA * kA * 2);
    unsigned short* wqm_t = (unsigned short*)take((size_t)kA * kA * 2);
    unsigned short* wkc_t = (unsigned short*)take((size_t)kA * kA * 2);
    unsigned short* wqc_t = (unsigned short*)take((size_t)kA * kA * 2);
    float* p_buf     = (float*)take((size_t)kB * kQ * kK * 4);
    float* cp_buf    = (float*)take((size_t)kB * kQ * kK * 4);
    float* alpha_buf = (float*)take((size_t)kB * kQ * kK * 4);

    // --- preprocessing: activations -> bf16, weights -> bf16 transposed
    const int nk4 = kB * kK * kA / 4;            // 2,097,152
    const int nq4 = kB * kQ * kA / 4;            // 262,144
    cvt_bf16<<<(nk4 + 255) / 256, 256, 0, stream>>>(key_enc, keyb, nk4);
    cvt_bf16<<<(nq4 + 255) / 256, 256, 0, stream>>>(query,   qryb, nq4);
    wtr_bf16<<<256, 256, 0, stream>>>(Wk_m, wkm_t);
    wtr_bf16<<<256, 256, 0, stream>>>(Wq_m, wqm_t);
    wtr_bf16<<<256, 256, 0, stream>>>(Wk_c, wkc_t);
    wtr_bf16<<<256, 256, 0, stream>>>(Wq_c, wqc_t);

    // --- projections (bf16 WMMA, bf16 outputs)
    const int tiles_k = (kB * kK / 16) * (kA / 16);  // 32768
    const int tiles_q = (kB * kQ / 16) * (kA / 16);  // 4096
    proj_bf16_wmma<<<tiles_k / 4, 128, 0, stream>>>(keyb, wkm_t, bk_m, km, kB * kK);
    proj_bf16_wmma<<<tiles_k / 4, 128, 0, stream>>>(keyb, wkc_t, bk_c, kc, kB * kK);
    proj_bf16_wmma<<<tiles_q / 4, 128, 0, stream>>>(qryb, wqm_t, bq_m, qm, kB * kQ);
    proj_bf16_wmma<<<tiles_q / 4, 128, 0, stream>>>(qryb, wqc_t, bq_c, qc, kB * kQ);

    // --- monotonic path
    const size_t lds2 = (size_t)16 * kK * sizeof(float);
    mono_pchoose_wmma<<<kB * (kQ / 16), 256, lds2, stream>>>(
            qm, km, noise, rscal, p_buf, cp_buf);

    alpha_scan<<<kB, 512, 0, stream>>>(p_buf, cp_buf, alpha_buf);

    // --- chunkwise path + beta
    const size_t lds4 = (size_t)16 * kK * sizeof(float) + 16 * sizeof(float);
    chunk_beta_wmma<<<kB * kH * (kQ / 16), 256, lds4, stream>>>(
            qc, kc, alpha_buf, out);
}
